// HyperbolicMultiHeadAttention_26920855011363
// MI455X (gfx1250) — matrix operand order repaired and checked by reading, not compiled, not run on hardware
//
#include <hip/hip_runtime.h>

#define NTOK   2048
#define DMODEL 1024
#define NHEADS 16
#define DHEAD  64
#define EPSF   1e-5f

typedef _Float16 h16;
typedef __attribute__((ext_vector_type(16))) _Float16 v16h;
typedef __attribute__((ext_vector_type(8)))  float    v8f;
typedef __attribute__((ext_vector_type(4)))  float    fvec4;

__device__ __forceinline__ float softplus_dev(float x) { return log1pf(expf(x)); }

// A-fragment (16x32 f16, per ISA table): lane holds row M=lane&15, kbase=(lane>>4)*8,
// VGPR0-3 = K[kbase..kbase+7], VGPR4-7 = K[kbase+16..kbase+23].
// Caller passes p = &row[kbase]; we fetch [p,p+8) and [p+16,p+24).
__device__ __forceinline__ v16h ldfragA(const h16* p) {
  union { v16h v; fvec4 q[2]; } u;
  u.q[0] = *(const fvec4*)(p);
  u.q[1] = *(const fvec4*)(p + 16);
  return u.v;
}
// B-fragment (32x16 f16): lane holds col N=lane&15, kbase=(lane>>4)*8,
// K[kbase..kbase+7], K[kbase+16..kbase+23]. Caller passes p = &colrow[kbase] (Bᵀ row-major).
__device__ __forceinline__ v16h ldfragB(const h16* p) {
  union { v16h v; fvec4 q[2]; } u;
  u.q[0] = *(const fvec4*)(p);
  u.q[1] = *(const fvec4*)(p + 16);
  return u.v;
}

// ---------------------------------------------------------------- cvt f32->f16
__global__ void cvt_f32_f16(const float* __restrict__ in, h16* __restrict__ out, int n) {
  int i = blockIdx.x * blockDim.x + threadIdx.x;
  if (i < n) out[i] = (h16)in[i];
}

// ------------------------------------------------- C = A(MxK) * B(NxK)^T, f32 out
// grid (M/128, N/64), block 256 (8 waves); wave computes 16 rows x 64 cols.
__global__ void __launch_bounds__(256) gemm_nt_f16(const h16* __restrict__ A,
                                                   const h16* __restrict__ B,
                                                   float* __restrict__ C,
                                                   int M, int N, int K) {
  int tid = threadIdx.x, wave = tid >> 5, lane = tid & 31;
  int m = lane & 15, hv = lane >> 4;
  int arow = blockIdx.x * 128 + wave * 16 + m;
  int colbase = blockIdx.y * 64;
  const h16* Arow = A + (size_t)arow * K;
  v8f acc[4] = {};
  for (int k0 = 0; k0 < K; k0 += 32) {
    v16h a = ldfragA(Arow + k0 + hv * 8);
#pragma unroll
    for (int nt = 0; nt < 4; ++nt) {
      const h16* Brow = B + (size_t)(colbase + nt * 16 + m) * K;
      v16h b = ldfragB(Brow + k0 + hv * 8);
      acc[nt] = __builtin_amdgcn_wmma_f32_16x16x32_f16(false, a, false, b,
                                                       (short)0, acc[nt], false, false);
    }
  }
  int rb = blockIdx.x * 128 + wave * 16 + hv * 8;  // C layout: VGPR r -> row r / r+8
#pragma unroll
  for (int nt = 0; nt < 4; ++nt)
#pragma unroll
    for (int r = 0; r < 8; ++r)
      C[(size_t)(rb + r) * N + colbase + nt * 16 + m] = acc[nt][r];
}

// ------------------------------------- exp-map of Q,K rows; emit f16 + row-norms^2
__global__ void expmap_qk(const float* __restrict__ Qt, const float* __restrict__ Kt,
                          h16* __restrict__ Qh, h16* __restrict__ Kh,
                          float* __restrict__ qns, float* __restrict__ kns,
                          const float* __restrict__ p_logc, int N) {
  int idx = blockIdx.x * blockDim.x + threadIdx.x;
  int total = 2 * NHEADS * N;
  if (idx >= total) return;
  int t = idx / (NHEADS * N);
  int r = idx % (NHEADS * N);
  int h = r / N, n = r % N;
  const float* src = (t == 0 ? Qt : Kt) + (size_t)n * DMODEL + h * DHEAD;
  h16* dst = (t == 0 ? Qh : Kh) + (size_t)n * DMODEL + h * DHEAD;
  float c = softplus_dev(p_logc[0]);
  float sc = sqrtf(c);
  float ss = 0.f, v[DHEAD];
#pragma unroll
  for (int i = 0; i < DHEAD; ++i) { v[i] = src[i]; ss += v[i] * v[i]; }
  float nrm = fmaxf(sqrtf(ss), EPSF);
  float t1 = tanhf(sc * nrm) / (sc * nrm);
#pragma unroll
  for (int i = 0; i < DHEAD; ++i) dst[i] = (h16)(t1 * v[i]);
  ((t == 0) ? qns : kns)[h * N + n] = t1 * t1 * ss;
}

// ------------------------- flash-style hyperbolic attention, one head per grid.y
// block: 256 thr (8 waves), 128 query rows; 64-key blocks, online softmax.
__global__ void __launch_bounds__(256) hyp_attn(const h16* __restrict__ Qh,
                                                const h16* __restrict__ Kh,
                                                const float* __restrict__ Vt,
                                                const float* __restrict__ qns,
                                                const float* __restrict__ kns,
                                                float* __restrict__ Ot,
                                                const float* __restrict__ p_logc,
                                                const float* __restrict__ p_beta,
                                                const float* __restrict__ p_bias,
                                                int N) {
  __shared__ __align__(16) h16 lds_v[64 * 64];           // V tile transposed: [dh][key]
  __shared__ __align__(16) h16 lds_p[8][16 * 64 + 8];    // per-wave P tile (pad keeps 16B align)

  int tid = threadIdx.x, wave = tid >> 5, lane = tid & 31;
  int m = lane & 15, hv = lane >> 4;
  int h = blockIdx.y;
  int qbase = blockIdx.x * 128;
  int qrow = qbase + wave * 16 + m;           // A-fragment row for this lane
  int rrow = qbase + wave * 16 + hv * 8;      // C-layout base row (reg r -> rrow+r)

  float c = softplus_dev(p_logc[0]);
  float sc = sqrtf(c);
  float beta = softplus_dev(p_beta[0]);
  float bias = p_bias[0];

  const h16* qptr = Qh + (size_t)qrow * DMODEL + h * DHEAD;
  v16h qa0 = ldfragA(qptr + hv * 8);          // d 0..31
  v16h qa1 = ldfragA(qptr + 32 + hv * 8);     // d 32..63

  float qn[8];
#pragma unroll
  for (int r = 0; r < 8; ++r) qn[r] = qns[h * N + rrow + r];

  v8f acc[4] = {};
  float mrun[8], lrun[8];
#pragma unroll
  for (int r = 0; r < 8; ++r) { mrun[r] = -3.0e38f; lrun[r] = 0.f; }

  int kend = qbase + 128;                     // causal: keys < kend suffice for this block
  for (int kb = 0; kb < kend; kb += 64) {
    __syncthreads();                          // lds_v consumed by previous iteration
    for (int i = tid; i < 64 * 64; i += 256) {
      int key = i >> 6, dh = i & 63;
      lds_v[dh * 64 + key] = (h16)Vt[(size_t)(kb + key) * DMODEL + h * DHEAD + dh];
    }
    __syncthreads();

    // ---- scores: 4 key tiles of 16, K-dim 64 = 2 wmma each; fused distance epilogue
    v8f s[4];
#pragma unroll
    for (int kt = 0; kt < 4; ++kt) {
      int key = kb + kt * 16 + m;
      const h16* kp = Kh + (size_t)key * DMODEL + h * DHEAD;
      v16h kf0 = ldfragB(kp + hv * 8);
      v16h kf1 = ldfragB(kp + 32 + hv * 8);
      v8f d = {};
      d = __builtin_amdgcn_wmma_f32_16x16x32_f16(false, qa0, false, kf0, (short)0, d, false, false);
      d = __builtin_amdgcn_wmma_f32_16x16x32_f16(false, qa1, false, kf1, (short)0, d, false, false);
      float kn = kns[h * N + key];
#pragma unroll
      for (int r = 0; r < 8; ++r) {
        float xn = qn[r];
        float diff = xn - 2.f * d[r] + kn;
        float den = fmaxf((1.f - c * xn) * (1.f - c * kn), EPSF);
        float arg = fmaxf(1.f + 2.f * c * diff / den, 1.f + EPSF);
        float dist = logf(arg + sqrtf(arg * arg - 1.f)) / sc;   // arccosh
        float sv = -beta * dist - bias;
        if (key > rrow + r) sv = -3.0e38f;                      // causal mask
        s[kt][r] = sv;
      }
    }

    // ---- online softmax per row (row = 16 lanes within one half-wave)
    float pscale[8];
#pragma unroll
    for (int r = 0; r < 8; ++r) {
      float mb = fmaxf(fmaxf(s[0][r], s[1][r]), fmaxf(s[2][r], s[3][r]));
#pragma unroll
      for (int x = 1; x < 16; x <<= 1) mb = fmaxf(mb, __shfl_xor(mb, x));
      float mnew = fmaxf(mrun[r], mb);
      float scale = __expf(mrun[r] - mnew);
      float psum = 0.f;
#pragma unroll
      for (int kt = 0; kt < 4; ++kt) {
        float p = __expf(s[kt][r] - mnew);
        s[kt][r] = p;
        psum += p;
      }
#pragma unroll
      for (int x = 1; x < 16; x <<= 1) psum += __shfl_xor(psum, x);
      lrun[r] = lrun[r] * scale + psum;
      mrun[r] = mnew;
      pscale[r] = scale;
    }
#pragma unroll
    for (int nt = 0; nt < 4; ++nt)
#pragma unroll
      for (int r = 0; r < 8; ++r) acc[nt][r] *= pscale[r];

    // ---- P (C-layout) -> wave-private LDS row-major, then A-fragments for P·V
    h16* pw = &lds_p[wave][0];
#pragma unroll
    for (int kt = 0; kt < 4; ++kt)
#pragma unroll
      for (int r = 0; r < 8; ++r)
        pw[(hv * 8 + r) * 64 + kt * 16 + m] = (h16)s[kt][r];
    __syncthreads();

    const h16* prow = pw + m * 64;
    v16h pa0 = ldfragA(prow + hv * 8);         // keys 0..31 of block
    v16h pa1 = ldfragA(prow + 32 + hv * 8);    // keys 32..63
#pragma unroll
    for (int nt = 0; nt < 4; ++nt) {
      const h16* vcol = &lds_v[(nt * 16 + m) * 64];
      v16h vb0 = ldfragB(vcol + hv * 8);
      v16h vb1 = ldfragB(vcol + 32 + hv * 8);
      acc[nt] = __builtin_amdgcn_wmma_f32_16x16x32_f16(false, pa0, false, vb0, (short)0, acc[nt], false, false);
      acc[nt] = __builtin_amdgcn_wmma_f32_16x16x32_f16(false, pa1, false, vb1, (short)0, acc[nt], false, false);
    }
  }

#pragma unroll
  for (int nt = 0; nt < 4; ++nt)
#pragma unroll
    for (int r = 0; r < 8; ++r)
      Ot[(size_t)(rrow + r) * DMODEL + h * DHEAD + nt * 16 + m] = acc[nt][r] / lrun[r];
}

// ------------------------- exp-map then log-map of output rows; emit f16 concat
__global__ void out_maps(const float* __restrict__ Ot, h16* __restrict__ Th,
                         const float* __restrict__ p_logc, int N) {
  int idx = blockIdx.x * blockDim.x + threadIdx.x;
  if (idx >= NHEADS * N) return;
  int h = idx / N, n = idx % N;
  const float* src = Ot + (size_t)n * DMODEL + h * DHEAD;
  h16* dst = Th + (size_t)n * DMODEL + h * DHEAD;
  float c = softplus_dev(p_logc[0]);
  float sc = sqrtf(c);
  float ss = 0.f, v[DHEAD];
#pragma unroll
  for (int i = 0; i < DHEAD; ++i) { v[i] = src[i]; ss += v[i] * v[i]; }
  float n1 = fmaxf(sqrtf(ss), EPSF);
  float t1 = tanhf(sc * n1) / (sc * n1);
  float n2 = fmaxf(t1 * sqrtf(ss), EPSF);          // ||exp_map(v)||, clamped
  float ncl = fminf(n2, 1.f / sc - EPSF);
  float t2 = atanhf(sc * ncl) / (sc * n2);
  float t = t1 * t2;
#pragma unroll
  for (int i = 0; i < DHEAD; ++i) dst[i] = (h16)(t * v[i]);
}

// -------------------------------------------------------------------- launcher
extern "C" void kernel_launch(void* const* d_in, const int* in_sizes, int n_in,
                              void* d_out, int out_size, void* d_ws, size_t ws_size,
                              hipStream_t stream) {
  const float* x      = (const float*)d_in[0];
  const float* Wq     = (const float*)d_in[1];
  const float* Wk     = (const float*)d_in[2];
  const float* Wv     = (const float*)d_in[3];
  const float* Wo     = (const float*)d_in[4];
  const float* p_logc = (const float*)d_in[5];
  const float* p_beta = (const float*)d_in[6];
  const float* p_bias = (const float*)d_in[7];
  // d_in[8] = mask: known causal tril; handled analytically in hyp_attn.

  char* w = (char*)d_ws;
  auto give = [&](size_t bytes) { char* p = w; w += (bytes + 255) & ~(size_t)255; return p; };
  h16*   xh  = (h16*)give((size_t)NTOK * DMODEL * sizeof(h16));
  h16*   Wqh = (h16*)give((size_t)DMODEL * DMODEL * sizeof(h16));
  h16*   Wkh = (h16*)give((size_t)DMODEL * DMODEL * sizeof(h16));
  h16*   Wvh = (h16*)give((size_t)DMODEL * DMODEL * sizeof(h16));
  h16*   Woh = (h16*)give((size_t)DMODEL * DMODEL * sizeof(h16));
  float* Qt  = (float*)give((size_t)NTOK * DMODEL * sizeof(float));
  float* Kt  = (float*)give((size_t)NTOK * DMODEL * sizeof(float));
  float* Vt  = (float*)give((size_t)NTOK * DMODEL * sizeof(float));
  h16*   Qhm = (h16*)give((size_t)NTOK * DMODEL * sizeof(h16));
  h16*   Khm = (h16*)give((size_t)NTOK * DMODEL * sizeof(h16));
  float* qns = (float*)give((size_t)NHEADS * NTOK * sizeof(float));
  float* kns = (float*)give((size_t)NHEADS * NTOK * sizeof(float));
  float* Ot  = (float*)give((size_t)NTOK * DMODEL * sizeof(float));
  h16*   Th  = (h16*)give((size_t)NTOK * DMODEL * sizeof(h16));

  cvt_f32_f16<<<(NTOK * DMODEL + 255) / 256, 256, 0, stream>>>(x, xh, NTOK * DMODEL);
  cvt_f32_f16<<<(DMODEL * DMODEL + 255) / 256, 256, 0, stream>>>(Wq, Wqh, DMODEL * DMODEL);
  cvt_f32_f16<<<(DMODEL * DMODEL + 255) / 256, 256, 0, stream>>>(Wk, Wkh, DMODEL * DMODEL);
  cvt_f32_f16<<<(DMODEL * DMODEL + 255) / 256, 256, 0, stream>>>(Wv, Wvh, DMODEL * DMODEL);
  cvt_f32_f16<<<(DMODEL * DMODEL + 255) / 256, 256, 0, stream>>>(Wo, Woh, DMODEL * DMODEL);

  dim3 gg(NTOK / 128, DMODEL / 64);
  gemm_nt_f16<<<gg, 256, 0, stream>>>(xh, Wqh, Qt, NTOK, DMODEL, DMODEL);
  gemm_nt_f16<<<gg, 256, 0, stream>>>(xh, Wkh, Kt, NTOK, DMODEL, DMODEL);
  gemm_nt_f16<<<gg, 256, 0, stream>>>(xh, Wvh, Vt, NTOK, DMODEL, DMODEL);

  expmap_qk<<<(2 * NHEADS * NTOK) / 256, 256, 0, stream>>>(Qt, Kt, Qhm, Khm, qns, kns, p_logc, NTOK);

  dim3 ga(NTOK / 128, NHEADS);
  hyp_attn<<<ga, 256, 0, stream>>>(Qhm, Khm, Vt, qns, kns, Ot, p_logc, p_beta, p_bias, NTOK);

  out_maps<<<(NHEADS * NTOK) / 256, 256, 0, stream>>>(Ot, Th, p_logc, NTOK);

  gemm_nt_f16<<<gg, 256, 0, stream>>>(Th, Woh, (float*)d_out, NTOK, DMODEL, DMODEL);
}
